// LGFM_26645977104432
// MI455X (gfx1250) — compile-verified, matrix operand order checked by reading, not run on hardware
//
#include <hip/hip_runtime.h>

// ---------------------------------------------------------------------------
// LGFM on MI455X (gfx1250, wave32).
// Dominant cost: 4 fused passes of A(8192x8192, fp32) @ Z(8192x256, bf16)
// using v_wmma_f32_16x16x32_bf16. A streams fp32 from HBM (1.07 GB total,
// ~46us at 23.3 TB/s): staged into LDS per-workgroup via the Tensor Data
// Mover (double-buffered, s_wait_tensorcnt), converted fp32->bf16 in
// registers. Z is pre-transposed (ZT[256][8192] bf16, 4MB, L2-resident) and
// its WMMA B-fragments are double-buffered in VGPRs (K loop unrolled x2 so
// buffer indices are literals; last two steps peeled so all pipeline flags
// are compile-time). Barriers are raw s_barrier_signal/wait pairs (no
// full-fence waitcnt) so in-flight B prefetches survive across them.
// ---------------------------------------------------------------------------

#define N_     8192
#define FEAT_  200
#define SEG_   50
#define NSEG_  4
#define ZCOLS_ 256            // 4 segments padded to 64 cols each
#define OUT_   300
#define SEGN_  (N_ * SEG_)    // elements in one N x 50 buffer
#define KTILE_ 32
#define LDSW_  36             // padded fp32 row stride (conflict-free, 16B-aligned)

typedef __attribute__((ext_vector_type(16))) __bf16 v16bf;
typedef __attribute__((ext_vector_type(8)))  __bf16 v8bf;
typedef __attribute__((ext_vector_type(8)))  float  v8f;

typedef unsigned int tdm_u32x4 __attribute__((ext_vector_type(4)));
typedef int          tdm_i32x8 __attribute__((ext_vector_type(8)));
typedef int          tdm_i32x4 __attribute__((ext_vector_type(4)));

#if defined(__has_builtin)
#if __has_builtin(__builtin_amdgcn_tensor_load_to_lds)
#define HAVE_TDM 1
#endif
#if __has_builtin(__builtin_amdgcn_global_load_async_to_lds_b128)
#define HAVE_ASYNC_LDS 1
#endif
#if __has_builtin(__builtin_amdgcn_s_wait_tensorcnt)
#define HAVE_WAIT_TENSOR 1
#endif
#if __has_builtin(__builtin_amdgcn_s_wait_asynccnt)
#define HAVE_WAIT_ASYNC 1
#endif
#endif

__device__ __forceinline__ void wait_tensor_le(int which) {  // 0/1 literal
#if HAVE_WAIT_TENSOR
  if (which) __builtin_amdgcn_s_wait_tensorcnt(1);
  else       __builtin_amdgcn_s_wait_tensorcnt(0);
#else
  if (which) asm volatile("s_wait_tensorcnt 0x1" ::: "memory");
  else       asm volatile("s_wait_tensorcnt 0x0" ::: "memory");
#endif
}

__device__ __forceinline__ void wait_async_le(int which) {
#if HAVE_WAIT_ASYNC
  if (which) __builtin_amdgcn_s_wait_asynccnt(1);
  else       __builtin_amdgcn_s_wait_asynccnt(0);
#else
  if (which) asm volatile("s_wait_asynccnt 0x1" ::: "memory");
  else       asm volatile("s_wait_asynccnt 0x0" ::: "memory");
#endif
}

// Raw workgroup split-barrier: compiler-level ordering only (no waitcnt
// fence), so outstanding global loads stay in flight across it.
__device__ __forceinline__ void wg_barrier_raw() {
  asm volatile("s_barrier_signal -1\n\ts_barrier_wait -1" ::: "memory");
}

// ---------------------------------------------------------------------------
// Big GEMM: for seg s in 0..3:  out[s] = relu(A @ Zseg + bias)
// WG = 256 threads = 8 waves covering a 32(M) x 256(N) tile; wave tile is
// 16(M) x 64(N) = 4 accumulators; K loop over 8192 in steps of 32.
// A tile (32x32 fp32) staged in LDS via TDM / async-LDS / sync fallback,
// double buffered with prefetch depth 2; B fragments double-buffered in VGPRs.
// ---------------------------------------------------------------------------
__global__ __launch_bounds__(256) void gcn_gemm_k(
    const float* __restrict__ A, const __bf16* __restrict__ ZT,
    const float* __restrict__ bias, float* __restrict__ outBase)
{
  __shared__ float Asmem[2][32 * LDSW_];   // 2 x 4.5 KB

  const int tid  = threadIdx.x;
  const int lane = tid & 31;
  const int w    = tid >> 5;
  const int mblk = blockIdx.x * 32;
  const int m0   = mblk + (w & 1) * 16;    // wave's 16-row strip
  const int n0b  = (w >> 1) * 64;          // wave's 64-col group
  const int khalf = (lane < 16) ? 0 : 8;   // K sub-block select (ISA frag layout)
  const int nloc = lane & 15;
  // staging map for fallback paths: thread -> (row, 4-float chunk)
  const int sr = tid >> 3, sk = (tid & 7) * 4;

  const __bf16* zt[4];
  #pragma unroll
  for (int t = 0; t < 4; ++t)
    zt[t] = ZT + (size_t)(n0b + t * 16 + nloc) * N_;

  // Issue staging of A block starting at column k0 into LDS buffer `buf`.
  auto stageA = [&](int buf, int k0) {
#if HAVE_TDM
    if (w == 0) {
      // Tensor DMA descriptor (D# groups 0/1), 2D tile 32(k) x 32(m) fp32,
      // pad 4 dwords after every 32 dwords -> LDS row stride = LDSW_.
      unsigned long long gaddr =
          (unsigned long long)(const void*)(A + (size_t)mblk * N_ + k0);
      unsigned int ldsoff =
          (unsigned int)(unsigned long long)(const void*)&Asmem[buf][0];
      tdm_u32x4 g0;
      g0[0] = 1u;                                       // count=1, user desc
      g0[1] = ldsoff;                                   // lds_addr (bytes)
      g0[2] = (unsigned int)gaddr;                      // global_addr[31:0]
      g0[3] = (unsigned int)((gaddr >> 32) & 0x01FFFFFFu) | (2u << 30); // type=2
      tdm_i32x8 g1;
      g1[0] = (2 << 16)      // data_size = 4B
            | (1 << 20)      // pad_enable
            | (4 << 22)      // pad_interval: every 32 dwords
            | (3 << 25);     // pad_amount: 4 dwords
      g1[1] = (int)(8192u << 16);  // tensor_dim0[15:0]  @ bits [63:48]
      g1[2] = (int)(8192u << 16);  // tensor_dim1[15:0]  @ bits [95:80]
      g1[3] = (int)(32u  << 16);   // tile_dim0 = 32     @ bits [127:112]
      g1[4] = 32;                  // tile_dim1 = 32     @ bits [143:128]
      g1[5] = 8192;                // tensor_dim0_stride[31:0]
      g1[6] = 0;
      g1[7] = 0;
      tdm_i32x4 gz4 = {0, 0, 0, 0};
      tdm_i32x8 gz8 = {0, 0, 0, 0, 0, 0, 0, 0};
      // 6-arg form (clang-23 / therock-10.0 headers): groups 2/3 unused (2D).
      __builtin_amdgcn_tensor_load_to_lds(g0, g1, gz4, gz4, gz8, 0);
    }
#elif HAVE_ASYNC_LDS
    __builtin_amdgcn_global_load_async_to_lds_b128(
        (const __attribute__((address_space(1))) void*)
            (A + (size_t)(mblk + sr) * N_ + k0 + sk),
        (__attribute__((address_space(3))) void*)&Asmem[buf][sr * LDSW_ + sk],
        0, 0);
#else
    uint4 tmp = *(const uint4*)(A + (size_t)(mblk + sr) * N_ + k0 + sk);
    *(uint4*)&Asmem[buf][sr * LDSW_ + sk] = tmp;
#endif
  };

  auto waitA = [&](int more_outstanding) {
#if HAVE_TDM
    if (w == 0) wait_tensor_le(more_outstanding);
#elif HAVE_ASYNC_LDS
    wait_async_le(more_outstanding);
#else
    (void)more_outstanding;
    asm volatile("s_wait_dscnt 0x0" ::: "memory");  // inter-wave ds_store vis
#endif
    wg_barrier_raw();
  };

  // B fragments (32x16 bf16 each), two register buffers of 4 fragments.
  v16bf bfrag[2][4];
  auto loadB = [&](int kb, int slot) {
    #pragma unroll
    for (int t = 0; t < 4; ++t) {
      union { v16bf v; v8bf h[2]; } ub;
      ub.h[0] = *(const v8bf*)(zt[t] + kb);
      ub.h[1] = *(const v8bf*)(zt[t] + kb + 16);
      bfrag[slot][t] = ub.v;
    }
  };

  v8f acc[4] = {};
  const int NIT = N_ / KTILE_;   // 256 (even)

  stageA(0, 0);
  stageA(1, KTILE_);
  loadB(khalf, 0);

  // One pipeline step. SLOT/MORE/DOB/DOSTAGE must be literals so the B-frag
  // buffers stay in VGPRs and the waits are single straight-line SOPPs.
  #define KSTEP(IDX, SLOT, MORE, DOB, DOSTAGE)                                \
  {                                                                           \
    const int k0_ = (IDX) * KTILE_;                                           \
    waitA(MORE);                          /* Asmem[SLOT] ready for all */     \
    const float* as_ = &Asmem[SLOT][((w & 1) * 16 + nloc) * LDSW_ + khalf];   \
    float4 f0_ = *(const float4*)(as_);                                       \
    float4 f1_ = *(const float4*)(as_ + 4);                                   \
    float4 f2_ = *(const float4*)(as_ + 16);                                  \
    float4 f3_ = *(const float4*)(as_ + 20);                                  \
    float af_[16] = {f0_.x, f0_.y, f0_.z, f0_.w, f1_.x, f1_.y, f1_.z, f1_.w,  \
                     f2_.x, f2_.y, f2_.z, f2_.w, f3_.x, f3_.y, f3_.z, f3_.w}; \
    union { v16bf v; __bf16 e[16]; } ua_;                                     \
    _Pragma("unroll")                                                         \
    for (int e = 0; e < 16; ++e) ua_.e[e] = (__bf16)af_[e];                   \
    if (DOB) loadB(k0_ + KTILE_ + khalf, (SLOT) ^ 1);                         \
    _Pragma("unroll")                                                         \
    for (int t = 0; t < 4; ++t)                                               \
      acc[t] = __builtin_amdgcn_wmma_f32_16x16x32_bf16(                       \
          false, ua_.v, false, bfrag[SLOT][t], (short)0, acc[t],              \
          false, false);                                                      \
    wg_barrier_raw();                     /* all waves done with Asmem[SLOT] */\
    if (DOSTAGE) stageA(SLOT, k0_ + 2 * KTILE_);                              \
  }

  for (int i = 0; i < NIT - 2; i += 2) {
    KSTEP(i, 0, 1, 1, 1);
    KSTEP(i + 1, 1, 1, 1, 1);
  }
  KSTEP(NIT - 2, 0, 1, 1, 0);   // last B prefetch, no further staging
  KSTEP(NIT - 1, 1, 0, 0, 0);   // drain: wait tensorcnt==0, no prefetches
  #undef KSTEP

  // Epilogue: bias + relu, drop pad columns (>=50), store per-segment.
  const int rext = m0 + khalf;
  #pragma unroll
  for (int t = 0; t < 4; ++t) {
    int col = n0b + t * 16 + nloc;
    int seg = col >> 6, cs = col & 63;
    if (cs < SEG_) {
      float bb = bias[cs];
      float* o = outBase + (size_t)seg * SEGN_ + cs;
      #pragma unroll
      for (int v = 0; v < 8; ++v) {
        float r = acc[t][v] + bb;
        o[(size_t)(rext + v) * SEG_] = r > 0.f ? r : 0.f;
      }
    }
  }
}

// ---------------------------------------------------------------------------
// Build ZT[n][k] = sum_j H_seg[k][j] * W[j][c]  (bf16 output, padded cols = 0)
// H element (seg,k,j) at H[seg*segOff + k*rowStride + j].
// ---------------------------------------------------------------------------
__global__ __launch_bounds__(256) void zt_build_k(
    const float* __restrict__ H, long long segOff, int rowStride,
    const float* __restrict__ W, __bf16* __restrict__ ZT)
{
  int gid = blockIdx.x * 256 + threadIdx.x;   // ZCOLS_ * N_ threads
  int k = gid & (N_ - 1);
  int n = gid >> 13;
  int seg = n >> 6, c = n & 63;
  float s = 0.f;
  if (c < SEG_) {
    const float* h = H + (size_t)seg * segOff + (size_t)k * rowStride;
    #pragma unroll 5
    for (int j = 0; j < SEG_; ++j) s += h[j] * W[j * SEG_ + c];
  }
  ZT[(size_t)n * N_ + k] = (__bf16)s;
}

// ---------------------------------------------------------------------------
// Level-1 elementwise: XF1 = a*x + b*gF ; XS1b = a*x + b*gS  (seg 2: 0.2/0.8)
// ---------------------------------------------------------------------------
__global__ __launch_bounds__(256) void lvl1_combine_k(
    const float* __restrict__ x, const float* __restrict__ gF,
    const float* __restrict__ gS, float* __restrict__ XF1,
    float* __restrict__ XS1b)
{
  int gid = blockIdx.x * 256 + threadIdx.x;   // NSEG_ * SEGN_ threads
  int seg = gid / SEGN_;
  int rc  = gid - seg * SEGN_;
  int r = rc / SEG_, c = rc - r * SEG_;
  float xa = (seg == 2) ? 0.2f : 1.0f;
  float gb = (seg == 2) ? 0.8f : 1.0f;
  float xv = x[(size_t)r * FEAT_ + seg * SEG_ + c];
  size_t idx = (size_t)seg * SEGN_ + rc;
  XF1[idx]  = xa * xv + gb * gF[idx];
  XS1b[idx] = xa * xv + gb * gS[idx];
}

// ---------------------------------------------------------------------------
// Gate step: out = h2 + sigmoid([h1 h2] @ Wg + bg) * h1   (Wg: 100 x 50)
// ---------------------------------------------------------------------------
__global__ __launch_bounds__(256) void gate_k(
    const float* __restrict__ h1, const float* __restrict__ h2,
    const float* __restrict__ Wg, const float* __restrict__ bg,
    float* __restrict__ outp)
{
  int gid = blockIdx.x * 256 + threadIdx.x;   // SEGN_ threads
  int r = gid / SEG_, c = gid - r * SEG_;
  const float* h1r = h1 + (size_t)r * SEG_;
  const float* h2r = h2 + (size_t)r * SEG_;
  float t = bg[c];
  #pragma unroll 5
  for (int j = 0; j < SEG_; ++j) {
    t += h1r[j] * Wg[j * SEG_ + c];
    t += h2r[j] * Wg[(SEG_ + j) * SEG_ + c];
  }
  float sg = 1.0f / (1.0f + __expf(-t));
  outp[gid] = h2[gid] + sg * h1[gid];
}

// out = 0.2*a + 0.8*b
__global__ __launch_bounds__(256) void cmb_k(
    const float* __restrict__ a, const float* __restrict__ b,
    float* __restrict__ o)
{
  int gid = blockIdx.x * 256 + threadIdx.x;
  o[gid] = 0.2f * a[gid] + 0.8f * b[gid];
}

// ---------------------------------------------------------------------------
// Projection: out = h_L @ proj_W + proj_b, h_L = 8 blocks of N_ x 50.
// ---------------------------------------------------------------------------
__global__ __launch_bounds__(256) void proj_k(
    const float* b0, const float* b1, const float* b2, const float* b3,
    const float* b4, const float* b5, const float* b6, const float* b7,
    const float* __restrict__ Wp, const float* __restrict__ bp,
    float* __restrict__ outp)
{
  int gid = blockIdx.x * 256 + threadIdx.x;   // N_ * OUT_ threads
  int r = gid / OUT_, o = gid - r * OUT_;
  const float* blks[8] = {b0, b1, b2, b3, b4, b5, b6, b7};
  float s = bp[o];
  #pragma unroll
  for (int b = 0; b < 8; ++b) {
    const float* h  = blks[b] + (size_t)r * SEG_;
    const float* wc = Wp + (size_t)(b * SEG_) * OUT_ + o;
    #pragma unroll 5
    for (int j = 0; j < SEG_; ++j) s += h[j] * wc[(size_t)j * OUT_];
  }
  outp[gid] = s;
}

// ---------------------------------------------------------------------------
extern "C" void kernel_launch(void* const* d_in, const int* in_sizes, int n_in,
                              void* d_out, int out_size, void* d_ws, size_t ws_size,
                              hipStream_t stream) {
  (void)in_sizes; (void)n_in; (void)out_size; (void)ws_size;
  const float* x       = (const float*)d_in[0];
  const float* A_S     = (const float*)d_in[1];
  const float* A_F     = (const float*)d_in[2];
  const float* gcn_W   = (const float*)d_in[3];
  const float* gcn_b   = (const float*)d_in[4];
  const float* gates_W = (const float*)d_in[5];   // (12, 100, 50)
  const float* gates_b = (const float*)d_in[6];   // (12, 50)
  const float* proj_W  = (const float*)d_in[7];   // (400, 300)
  const float* proj_b  = (const float*)d_in[8];   // (300,)
  float* out = (float*)d_out;

  // Workspace bump allocator (256B aligned slices).
  size_t off = 0;
  auto alloc = [&](size_t bytes) -> char* {
    off = (off + 255) & ~(size_t)255;
    char* p = (char*)d_ws + off;
    off += bytes;
    return p;
  };
  const size_t ZTB  = (size_t)ZCOLS_ * N_ * sizeof(__bf16);   // 4 MB
  const size_t SEGB = (size_t)SEGN_ * sizeof(float);          // 1.6 MB

  __bf16* ZTx  = (__bf16*)alloc(ZTB);
  __bf16* ZT2F = (__bf16*)alloc(ZTB);
  __bf16* ZT2S = (__bf16*)alloc(ZTB);
  float* gF    = (float*)alloc(4 * SEGB);   // gcn(x_i, A_F)  (raw)
  float* gS    = (float*)alloc(4 * SEGB);   // gcn(x_i, A_S)  (raw)
  float* XF1   = (float*)alloc(4 * SEGB);   // t_i_XF1
  float* XS1b  = (float*)alloc(4 * SEGB);   // t_i_XS1b
  float* G2F   = (float*)alloc(4 * SEGB);   // gcn(t_i_XF1, A_F)
  float* G2S   = (float*)alloc(4 * SEGB);   // gcn(t_i_XS1b, A_S)
  float* cS    = (float*)alloc(4 * SEGB);   // gated XS1 chain (slots 1..3)
  float* cF    = (float*)alloc(4 * SEGB);   // gated XF1b chain (slots 1..3)
  float* XF2   = (float*)alloc(4 * SEGB);
  float* XS2   = (float*)alloc(4 * SEGB);
  float* P1    = (float*)alloc(SEGB);
  float* P2    = (float*)alloc(SEGB);

  const dim3 B(256);
  const int G_GEMM = N_ / 32;                    // 256 workgroups
  const int G_ZT   = (ZCOLS_ * N_) / 256;        // 8192
  const int G_E1   = (NSEG_ * SEGN_) / 256;      // 6400
  const int G_SEG  = SEGN_ / 256;                // 1600
  const int G_PRJ  = (N_ * OUT_) / 256;          // 9600

  // -------- level 1 --------
  zt_build_k<<<G_ZT, B, 0, stream>>>(x, (long long)SEG_, FEAT_, gcn_W, ZTx);
  gcn_gemm_k<<<G_GEMM, B, 0, stream>>>(A_F, ZTx, gcn_b, gF);
  gcn_gemm_k<<<G_GEMM, B, 0, stream>>>(A_S, ZTx, gcn_b, gS);
  lvl1_combine_k<<<G_E1, B, 0, stream>>>(x, gF, gS, XF1, XS1b);

  // -------- level 2 --------
  zt_build_k<<<G_ZT, B, 0, stream>>>(XF1,  (long long)SEGN_, SEG_, gcn_W, ZT2F);
  zt_build_k<<<G_ZT, B, 0, stream>>>(XS1b, (long long)SEGN_, SEG_, gcn_W, ZT2S);
  gcn_gemm_k<<<G_GEMM, B, 0, stream>>>(A_F, ZT2F, gcn_b, G2F);
  gcn_gemm_k<<<G_GEMM, B, 0, stream>>>(A_S, ZT2S, gcn_b, G2S);

  // Gate index map: GS1..3 = 0..2, GF1..3 = 3..5, GS1B..3B = 6..8, GF1B..3B = 9..11
  auto GW = [&](int g) { return gates_W + (size_t)g * 100 * SEG_; };
  auto GB = [&](int g) { return gates_b + (size_t)g * SEG_; };

  // chainS (gated t_i_XS1): slot 0 aliases raw gS[0]
  const float* cSp[4] = {gS, cS + 1 * SEGN_, cS + 2 * SEGN_, cS + 3 * SEGN_};
  for (int s = 1; s < 4; ++s)
    gate_k<<<G_SEG, B, 0, stream>>>(cSp[s - 1], gS + (size_t)s * SEGN_,
                                    GW(s - 1), GB(s - 1), (float*)cSp[s]);
  // chainFb (gated t_i_XF1b): slot 0 aliases raw gF[0]
  const float* cFp[4] = {gF, cF + 1 * SEGN_, cF + 2 * SEGN_, cF + 3 * SEGN_};
  for (int s = 1; s < 4; ++s)
    gate_k<<<G_SEG, B, 0, stream>>>(cFp[s - 1], gF + (size_t)s * SEGN_,
                                    GW(9 + s - 1), GB(9 + s - 1), (float*)cFp[s]);

  // XF2 chain: pre = 0.2*chainS + 0.8*G2F ; gated with GF1..GF3
  float* XF2p[4] = {XF2, XF2 + SEGN_, XF2 + 2 * SEGN_, XF2 + 3 * SEGN_};
  cmb_k<<<G_SEG, B, 0, stream>>>(cSp[0], G2F, XF2p[0]);
  for (int s = 1; s < 4; ++s) {
    cmb_k<<<G_SEG, B, 0, stream>>>(cSp[s], G2F + (size_t)s * SEGN_, P1);
    gate_k<<<G_SEG, B, 0, stream>>>(XF2p[s - 1], P1, GW(3 + s - 1), GB(3 + s - 1), XF2p[s]);
  }
  // XS2 chain: pre = 0.2*chainFb + 0.8*G2S ; gated with GS1B..GS3B
  float* XS2p[4] = {XS2, XS2 + SEGN_, XS2 + 2 * SEGN_, XS2 + 3 * SEGN_};
  cmb_k<<<G_SEG, B, 0, stream>>>(cFp[0], G2S, XS2p[0]);
  for (int s = 1; s < 4; ++s) {
    cmb_k<<<G_SEG, B, 0, stream>>>(cFp[s], G2S + (size_t)s * SEGN_, P2);
    gate_k<<<G_SEG, B, 0, stream>>>(XS2p[s - 1], P2, GW(6 + s - 1), GB(6 + s - 1), XS2p[s]);
  }

  // -------- projection --------
  proj_k<<<G_PRJ, B, 0, stream>>>(XF2p[0], XS2p[0], XF2p[1], XS2p[1],
                                  XF2p[2], XS2p[2], XF2p[3], XS2p[3],
                                  proj_W, proj_b, out);
}
